// ConditionalModel_62277025792832
// MI455X (gfx1250) — matrix operand order repaired and checked by reading, not compiled, not run on hardware
//
#include <hip/hip_runtime.h>

#define DEV __device__ __forceinline__

typedef __attribute__((ext_vector_type(16))) __bf16 bf16x16;
typedef __attribute__((ext_vector_type(8)))  __bf16 bf16x8;
typedef __attribute__((ext_vector_type(8)))  float  floatx8;

// ---- problem constants ----
constexpr int T_ = 800, B_ = 32, U_ = 64, V_ = 78, H_ = 400, KW_ = 10;
constexpr int HEADW = 121;
constexpr int K1  = 512;    // xcat1 padded width  [x(3) | w(78) | h1(400)] = 481 -> 512
constexpr int K23 = 896;    // xcat2/3 padded width [x(3) | h(400) | w(78) | hrec(400)] = 881 -> 896
constexpr int KH  = 1216;   // ycat padded width (3H=1200 -> 1216)
constexpr int KT1 = 16, KT23 = 28, KTA = 13, KTH = 38;
constexpr int NT  = 100;    // 1600/16 output col-tiles for LSTM GEMMs
constexpr int NWG = 25, NTHR = 256, GT = NWG * NTHR;  // 200 waves, 6400 threads
constexpr long ROWS = (long)B_ * T_;                  // 25600

struct Params {
  const float* x; const int* ch;
  const float *b1, *b2, *b3, *babk, *bhead;
  const __bf16 *pW1, *pW2, *pW3, *pWa, *pWh;
  __bf16 *xc1, *xc2, *xc3, *ycat;
  float *z, *c1, *c2, *c3, *kappa, *zhead, *out;
  unsigned* bar;
};

DEV float sigf(float x) { return 1.0f / (1.0f + expf(-x)); }

// grid-wide sense barrier (bar[0]=counter, bar[1]=generation)
DEV void gbar(unsigned* bar) {
  __syncthreads();
  if (threadIdx.x == 0) {
    volatile unsigned* vgen = bar + 1;
    unsigned g = *vgen;
    __threadfence();
    if (atomicAdd(bar, 1u) == (unsigned)(NWG - 1)) {
      atomicExch(bar, 0u);
      __threadfence();
      atomicAdd(bar + 1, 1u);
    } else {
      while (*vgen == g) __builtin_amdgcn_s_sleep(2);
    }
  }
  __syncthreads();
  __threadfence();
}

// A fragment (16x32 bf16): lane holds row M=lane&15, K runs [ks,ks+8) and [ks+16,ks+24)
DEV bf16x16 load_a(const __bf16* rowbase, int kt, int lane) {
  const __bf16* p = rowbase + kt * 32 + ((lane & 16) ? 8 : 0);
  bf16x8 lo = *(const bf16x8*)p;
  bf16x8 hi = *(const bf16x8*)(p + 16);
  bf16x16 a;
#pragma unroll
  for (int e = 0; e < 8; ++e) { a[e] = lo[e]; a[e + 8] = hi[e]; }
  return a;
}

DEV void store_d(float* z, int zstride, int m0, int n0, int lane, floatx8 acc) {
  int col   = n0 + (lane & 15);
  int rbase = m0 + ((lane & 16) ? 8 : 0);
#pragma unroll
  for (int j = 0; j < 8; ++j) z[(size_t)(rbase + j) * zstride + col] = acc[j];
}

// one 16x16 output tile: D = A(16xK) * W^T-tile, K = KT*32 (weights are L2-hot; no prefetch)
DEV void gemm_tile(const __bf16* xcat, int kstride, const __bf16* pW, int KT,
                   int mt, int nt, int lane, float* z, int zstride) {
  const __bf16* arow  = xcat + (size_t)(mt * 16 + (lane & 15)) * kstride;
  const __bf16* bbase = pW + (((size_t)nt * KT) * 32 + lane) * 16;
  floatx8 acc = {};
  for (int kt = 0; kt < KT; ++kt) {
    bf16x16 a = load_a(arow, kt, lane);
    bf16x16 b = *(const bf16x16*)(bbase + (size_t)kt * 32 * 16);
    acc = __builtin_amdgcn_wmma_f32_16x16x32_bf16(false, a, false, b, (short)0,
                                                  acc, false, false);
  }
  store_d(z, zstride, mt * 16, nt * 16, lane, acc);
}

// ---------------- init: zero state, build initial xcats (w=1, h=0, x=x[0]) ----------------
__global__ void kInit(unsigned* bar, float* c1, float* c2, float* c3, float* kappa,
                      __bf16* xc1, __bf16* xc2, __bf16* xc3, __bf16* ycat, const float* x) {
  const long s0 = 64, s1 = 3L * B_ * H_, s2 = B_ * KW_;
  const long s3 = B_ * K1, s4 = B_ * K23, s5 = B_ * K23, s6 = ROWS * 16;
  const long total = s0 + s1 + s2 + s3 + s4 + s5 + s6;
  for (long i = (long)blockIdx.x * blockDim.x + threadIdx.x; i < total;
       i += (long)gridDim.x * blockDim.x) {
    long li = i;
    if (li < s0) { bar[li] = 0u; continue; }
    li -= s0;
    if (li < s1) {
      if (li < 12800) c1[li] = 0.0f;
      else if (li < 25600) c2[li - 12800] = 0.0f;
      else c3[li - 25600] = 0.0f;
      continue;
    }
    li -= s1;
    if (li < s2) { kappa[li] = 0.0f; continue; }
    li -= s2;
    if (li < s3) {
      int b = (int)(li / K1), col = (int)(li % K1);
      float v = 0.0f;
      if (col < 3) v = x[b * 3 + col];          // x[t=0]
      else if (col < 81) v = 1.0f;              // w init = ones
      xc1[li] = (__bf16)v; continue;
    }
    li -= s3;
    if (li < s4) {
      int b = (int)(li / K23), col = (int)(li % K23);
      float v = 0.0f;
      if (col < 3) v = x[b * 3 + col];
      else if (col >= 403 && col < 481) v = 1.0f;
      xc2[li] = (__bf16)v; continue;
    }
    li -= s4;
    if (li < s5) {
      int b = (int)(li / K23), col = (int)(li % K23);
      float v = 0.0f;
      if (col < 3) v = x[b * 3 + col];
      else if (col >= 403 && col < 481) v = 1.0f;
      xc3[li] = (__bf16)v; continue;
    }
    li -= s5;
    { long r = li >> 4; int j = (int)(li & 15); ycat[r * KH + 1200 + j] = (__bf16)0.0f; }
  }
}

// ---------------- pack fp32 weights into WMMA B-fragment bf16 layout ----------------
__global__ void kPack(const float* W1ih, const float* W1hh, const float* W2ih, const float* W2hh,
                      const float* W3ih, const float* W3hh, const float* Wabk, const float* Whd,
                      __bf16* p1, __bf16* p2, __bf16* p3, __bf16* pa, __bf16* ph) {
  const long n1 = (long)NT * KT1 * 512, n23 = (long)NT * KT23 * 512;
  const long na = 2L * KTA * 512, nh = 8L * KTH * 512;
  const long total = n1 + 2 * n23 + na + nh;
  for (long i = (long)blockIdx.x * blockDim.x + threadIdx.x; i < total;
       i += (long)gridDim.x * blockDim.x) {
    long li = i; int which;
    if (li < n1) which = 0;
    else if ((li -= n1) < n23) which = 1;
    else if ((li -= n23) < n23) which = 2;
    else if ((li -= n23) < na) which = 3;
    else { li -= na; which = 4; }
    int KT = (which == 0) ? KT1 : (which <= 2) ? KT23 : (which == 3) ? KTA : KTH;
    long tile = li >> 9; int within = (int)(li & 511);
    int lane = within >> 4, e = within & 15;
    int nt = (int)(tile / KT), kt = (int)(tile % KT);
    int n = nt * 16 + (lane & 15);
    int k = kt * 32 + ((lane & 16) ? 8 : 0) + (e < 8 ? e : e + 8);
    float v = 0.0f;
    if (which == 0) { if (k < 81) v = W1ih[n * 81 + k]; else if (k < 481) v = W1hh[n * 400 + (k - 81)]; }
    else if (which == 1) { if (k < 481) v = W2ih[n * 481 + k]; else if (k < 881) v = W2hh[n * 400 + (k - 481)]; }
    else if (which == 2) { if (k < 481) v = W3ih[n * 481 + k]; else if (k < 881) v = W3hh[n * 400 + (k - 481)]; }
    else if (which == 3) { if (n < 30 && k >= 3 && k < 403) v = Wabk[n * 400 + (k - 3)]; }
    else { if (n < HEADW && k < 1200) v = Whd[n * 1200 + k]; }
    __bf16 bv = (__bf16)v;
    if (which == 0) p1[li] = bv;
    else if (which == 1) p2[li] = bv;
    else if (which == 2) p3[li] = bv;
    else if (which == 3) pa[li] = bv;
    else ph[li] = bv;
  }
}

// ---------------- persistent main kernel: recurrence + head ----------------
__global__ void __launch_bounds__(NTHR) kMain(Params p) {
  const int tid = threadIdx.x;
  const int lane = tid & 31;
  const int wid = tid >> 5;
  const int gwave = blockIdx.x * 8 + wid;     // 0..199
  const int gthr = blockIdx.x * NTHR + tid;   // 0..6399

  __shared__ float sZab[B_ * 32];             // attention logits (WG0 only)
  __shared__ float sAl[B_ * KW_];
  __shared__ float sBe[B_ * KW_];
  __shared__ float sPhi[B_ * U_];

  for (int t = 0; t < T_; ++t) {
    gbar(p.bar);  // xcats for step t ready (x_t, w_{t-1}, h_{t-1})

    // ---- LSTM1 GEMM: z(32x1600) = xcat1(32x512) @ W1^T, one tile per wave ----
    gemm_tile(p.xc1, K1, p.pW1, KT1, gwave & 1, gwave >> 1, lane, p.z, 1600);
    gbar(p.bar);

    // ---- act1: c1,h1 update; scatter h1 into xc1 (next step), xc2, ycat ----
    for (int i = gthr; i < B_ * H_; i += GT) {
      int b = i / H_, n = i % H_;
      const float* zr = p.z + b * 1600;
      float zi = zr[n] + p.b1[n];
      float zf = zr[n + 400] + p.b1[n + 400];
      float zg = zr[n + 800] + p.b1[n + 800];
      float zo = zr[n + 1200] + p.b1[n + 1200];
      float c = sigf(zf) * p.c1[i] + sigf(zi) * tanhf(zg);
      p.c1[i] = c;
      float h = sigf(zo) * tanhf(c);
      __bf16 hb = (__bf16)h;
      p.xc1[b * K1 + 81 + n] = hb;
      p.xc2[b * K23 + 3 + n] = hb;
      p.ycat[((size_t)b * T_ + t) * KH + n] = hb;
    }
    gbar(p.bar);

    // ---- attention: logits GEMM into LDS + window, all inside WG0 (no grid barrier) ----
    if (blockIdx.x == 0) {
      if (wid < 4)   // 4 tile-waves: zabk(32x32) = xcat2 @ Wabk^T (packed over h1 cols only)
        gemm_tile(p.xc2, K23, p.pWa, KTA, wid & 1, wid >> 1, lane, sZab, 32);
      __syncthreads();
      for (int i = tid; i < B_ * 30; i += NTHR) {
        int b = i / 30, j = i % 30;
        float v = expf(sZab[b * 32 + j] + p.babk[j]);
        if (j < 10) sAl[b * 10 + j] = v;
        else if (j < 20) sBe[b * 10 + (j - 10)] = v;
        else p.kappa[b * 10 + (j - 20)] += v;    // kappa += dk
      }
      __syncthreads();
      for (int i = tid; i < B_ * U_; i += NTHR) {
        int b = i >> 6, u = i & 63;
        float s = 0.0f;
#pragma unroll
        for (int k = 0; k < 10; ++k) {
          float d = p.kappa[b * 10 + k] - (float)u;
          s += sAl[b * 10 + k] * expf(-sBe[b * 10 + k] * d * d);
        }
        sPhi[i] = s;
      }
      __syncthreads();
      for (int i = tid; i < B_ * V_; i += NTHR) {
        int b = i / V_, v = i % V_;
        const int* cb = p.ch + b * U_;
        const float* ph = sPhi + b * U_;
        float s = 0.0f;
        for (int u = 0; u < U_; ++u) s += (cb[u] == v) ? ph[u] : 0.0f;
        __bf16 wb = (__bf16)s;
        p.xc1[b * K1 + 3 + v] = wb;      // used at t+1
        p.xc2[b * K23 + 403 + v] = wb;   // used by LSTM2 now
        p.xc3[b * K23 + 403 + v] = wb;   // used by LSTM3 now
      }
    }
    gbar(p.bar);

    // ---- LSTM2 GEMM ----
    gemm_tile(p.xc2, K23, p.pW2, KT23, gwave & 1, gwave >> 1, lane, p.z, 1600);
    gbar(p.bar);

    // ---- act2 ----
    for (int i = gthr; i < B_ * H_; i += GT) {
      int b = i / H_, n = i % H_;
      const float* zr = p.z + b * 1600;
      float zi = zr[n] + p.b2[n];
      float zf = zr[n + 400] + p.b2[n + 400];
      float zg = zr[n + 800] + p.b2[n + 800];
      float zo = zr[n + 1200] + p.b2[n + 1200];
      float c = sigf(zf) * p.c2[i] + sigf(zi) * tanhf(zg);
      p.c2[i] = c;
      float h = sigf(zo) * tanhf(c);
      __bf16 hb = (__bf16)h;
      p.xc2[b * K23 + 481 + n] = hb;   // h2 recurrent (t+1)
      p.xc3[b * K23 + 3 + n] = hb;     // LSTM3 input now
      p.ycat[((size_t)b * T_ + t) * KH + 400 + n] = hb;
    }
    gbar(p.bar);

    // ---- LSTM3 GEMM ----
    gemm_tile(p.xc3, K23, p.pW3, KT23, gwave & 1, gwave >> 1, lane, p.z, 1600);
    gbar(p.bar);

    // ---- act3 + stage x_{t+1} ----
    for (int i = gthr; i < B_ * H_; i += GT) {
      int b = i / H_, n = i % H_;
      const float* zr = p.z + b * 1600;
      float zi = zr[n] + p.b3[n];
      float zf = zr[n + 400] + p.b3[n + 400];
      float zg = zr[n + 800] + p.b3[n + 800];
      float zo = zr[n + 1200] + p.b3[n + 1200];
      float c = sigf(zf) * p.c3[i] + sigf(zi) * tanhf(zg);
      p.c3[i] = c;
      float h = sigf(zo) * tanhf(c);
      __bf16 hb = (__bf16)h;
      p.xc3[b * K23 + 481 + n] = hb;   // h3 recurrent (t+1)
      p.ycat[((size_t)b * T_ + t) * KH + 800 + n] = hb;
    }
    if (t + 1 < T_ && gthr < B_ * 3) {
      int b = gthr / 3, j = gthr % 3;
      __bf16 xv = (__bf16)p.x[((size_t)(t + 1) * B_ + b) * 3 + j];
      p.xc1[b * K1 + j] = xv;
      p.xc2[b * K23 + j] = xv;
      p.xc3[b * K23 + j] = xv;
    }
  }
  gbar(p.bar);  // ycat complete

  // ---- head GEMM: zhead(25600x128) = ycat(25600x1216) @ Whead^T ----
  for (int tile = gwave; tile < 1600 * 8; tile += 200) {
    int mt = tile >> 3, nt = tile & 7;
    const __bf16* arow  = p.ycat + (size_t)(mt * 16 + (lane & 15)) * KH;
    const __bf16* bbase = p.pWh + (((size_t)nt * KTH) * 32 + lane) * 16;
    floatx8 acc = {};
    for (int kt = 0; kt < KTH; ++kt) {
      if (kt + 1 < KTH) __builtin_prefetch(arow + (kt + 1) * 32, 0, 1);
      bf16x16 a = load_a(arow, kt, lane);
      bf16x16 b = *(const bf16x16*)(bbase + (size_t)kt * 32 * 16);
      acc = __builtin_amdgcn_wmma_f32_16x16x32_bf16(false, a, false, b, (short)0,
                                                    acc, false, false);
    }
    int col = nt * 16 + (lane & 15);
    float bias = (col < HEADW) ? p.bhead[col] : 0.0f;
    int rbase = mt * 16 + ((lane & 16) ? 8 : 0);
#pragma unroll
    for (int j = 0; j < 8; ++j) p.zhead[(size_t)(rbase + j) * 128 + col] = acc[j] + bias;
  }
  gbar(p.bar);

  // ---- head nonlinearities -> d_out, one (b,t) row per thread-iteration ----
  for (long r = gthr; r < ROWS; r += GT) {
    const float* zr = p.zhead + (size_t)r * 128;
    float* o = p.out + (size_t)r * HEADW;
    o[0] = 1.0f / (1.0f + expf(zr[0]));                   // sigmoid(-z0)
    float m = zr[1] * 4.0f;
    for (int k = 1; k < 20; ++k) m = fmaxf(m, zr[1 + k] * 4.0f);
    float s = 0.0f;
    for (int k = 0; k < 20; ++k) s += expf(zr[1 + k] * 4.0f - m);
    float inv = 1.0f / s;
    for (int k = 0; k < 20; ++k) o[1 + k] = expf(zr[1 + k] * 4.0f - m) * inv;   // pi softmax
    for (int k = 0; k < 20; ++k) o[21 + k] = zr[21 + k];                        // mu1
    for (int k = 0; k < 20; ++k) o[41 + k] = expf(zr[41 + k] - 3.0f);           // sig1
    for (int k = 0; k < 20; ++k) o[61 + k] = zr[61 + k];                        // mu2
    for (int k = 0; k < 20; ++k) o[81 + k] = expf(zr[81 + k] - 3.0f);           // sig2
    for (int k = 0; k < 20; ++k) o[101 + k] = tanhf(zr[101 + k]);               // ro
  }
}

extern "C" void kernel_launch(void* const* d_in, const int* in_sizes, int n_in,
                              void* d_out, int out_size, void* d_ws, size_t ws_size,
                              hipStream_t stream) {
  (void)in_sizes; (void)n_in; (void)out_size; (void)ws_size;
  const float* x    = (const float*)d_in[0];
  const int*   ch   = (const int*)d_in[1];
  const float* W1ih = (const float*)d_in[2];
  const float* W1hh = (const float*)d_in[3];
  const float* b1   = (const float*)d_in[4];
  const float* W2ih = (const float*)d_in[5];
  const float* W2hh = (const float*)d_in[6];
  const float* b2   = (const float*)d_in[7];
  const float* W3ih = (const float*)d_in[8];
  const float* W3hh = (const float*)d_in[9];
  const float* b3   = (const float*)d_in[10];
  const float* Wabk = (const float*)d_in[11];
  const float* babk = (const float*)d_in[12];
  const float* Whd  = (const float*)d_in[13];
  const float* bhd  = (const float*)d_in[14];

  char* w = (char*)d_ws;
  auto alloc = [&](size_t bytes) {
    char* p = w; w += (bytes + 255) & ~(size_t)255; return p;
  };
  unsigned* bar  = (unsigned*)alloc(256);
  __bf16* pW1    = (__bf16*)alloc((size_t)NT * KT1 * 512 * 2);
  __bf16* pW2    = (__bf16*)alloc((size_t)NT * KT23 * 512 * 2);
  __bf16* pW3    = (__bf16*)alloc((size_t)NT * KT23 * 512 * 2);
  __bf16* pWa    = (__bf16*)alloc((size_t)2 * KTA * 512 * 2);
  __bf16* pWh    = (__bf16*)alloc((size_t)8 * KTH * 512 * 2);
  __bf16* xc1    = (__bf16*)alloc((size_t)B_ * K1 * 2);
  __bf16* xc2    = (__bf16*)alloc((size_t)B_ * K23 * 2);
  __bf16* xc3    = (__bf16*)alloc((size_t)B_ * K23 * 2);
  float*  z      = (float*)alloc((size_t)B_ * 1600 * 4);
  float*  c1     = (float*)alloc((size_t)B_ * H_ * 4);
  float*  c2     = (float*)alloc((size_t)B_ * H_ * 4);
  float*  c3     = (float*)alloc((size_t)B_ * H_ * 4);
  float*  kappa  = (float*)alloc((size_t)B_ * KW_ * 4);
  __bf16* ycat   = (__bf16*)alloc((size_t)ROWS * KH * 2);
  float*  zhead  = (float*)alloc((size_t)ROWS * 128 * 4);

  kInit<<<256, 256, 0, stream>>>(bar, c1, c2, c3, kappa, xc1, xc2, xc3, ycat, x);
  kPack<<<1024, 256, 0, stream>>>(W1ih, W1hh, W2ih, W2hh, W3ih, W3hh, Wabk, Whd,
                                  pW1, pW2, pW3, pWa, pWh);

  Params p;
  p.x = x; p.ch = ch;
  p.b1 = b1; p.b2 = b2; p.b3 = b3; p.babk = babk; p.bhead = bhd;
  p.pW1 = pW1; p.pW2 = pW2; p.pW3 = pW3; p.pWa = pWa; p.pWh = pWh;
  p.xc1 = xc1; p.xc2 = xc2; p.xc3 = xc3; p.ycat = ycat;
  p.z = z; p.c1 = c1; p.c2 = c2; p.c3 = c3;
  p.kappa = kappa; p.zhead = zhead; p.out = (float*)d_out;
  p.bar = bar;

  kMain<<<NWG, NTHR, 0, stream>>>(p);
}
